// AudioEncoder_70325794504754
// MI455X (gfx1250) — matrix operand order repaired and checked by reading, not compiled, not run on hardware
//
#include <hip/hip_runtime.h>
#include <hip/hip_bf16.h>

// ---------------------------------------------------------------------------
// MI455X (gfx1250) Whisper-encoder. Compute-bound (~426 GF vs ~170MB weights =
// ~7us of HBM at 23.3 TB/s), so all large GEMMs run on v_wmma_f32_16x16x32_f16
// (wave32, f32 accumulate, f16 operands staged in LDS). Attention is a fused
// flash-style kernel (no O(T^2) buffer). CDNA5 data movers exercised:
//   - TENSOR_LOAD_TO_LDS (TDM) double-buffers the GEMM A tile (TENSORcnt)
//   - GLOBAL_LOAD_ASYNC_TO_LDS_B128 stages attention K tiles (ASYNCcnt)
//   - explicit s_wait_dscnt / s_wait_asynccnt / s_wait_tensorcnt
// ---------------------------------------------------------------------------

typedef __attribute__((ext_vector_type(16))) _Float16 v16h;
typedef __attribute__((ext_vector_type(8)))  _Float16 v8h;
typedef __attribute__((ext_vector_type(8)))  float    v8f;
typedef __attribute__((ext_vector_type(4)))  unsigned int v4u;
typedef __attribute__((ext_vector_type(8)))  int      v8i;
typedef __attribute__((ext_vector_type(4)))  int      v4i;

union U16 { v16h v; v8h h[2]; };

#define NB    4
#define NMELS 80
#define TIN   2048
#define TT    1024
#define EE    768
#define HH    12
#define HDD   64
#define LL    6
#define MFF   3072
#define CC    527

__device__ __forceinline__ float gelu_exact(float x) {
  return 0.5f * x * (1.0f + erff(x * 0.70710678118654752f));
}

// Issue a TDM load of a (128 rows x 32 cols) f16 tile (row stride = K elems)
// into LDS at lds_off. D# layout per CDNA5 ISA:
// group0: count=1 | lds_addr | global_addr | type=2 ("image")
// group1: data_size=1(2B), tensor_dim0/1 = 1<<24 (no clipping; exact tiles),
//         tile_dim0=32, tile_dim1=128, tensor_dim0_stride=K. groups 2/3 = 0.
// amdgpu-toolchain (clang-23) builtin: 6 args (g0,g1,g2,g3,g4,cpol).
__device__ __forceinline__ void tdm_load_tile(const _Float16* gp,
                                              unsigned lds_off, const v8i& g1) {
  unsigned long long ga = (unsigned long long)(uintptr_t)gp;
  v4u g0;
  g0[0] = 1u;                                     // count=1, user mode
  g0[1] = lds_off;                                // lds_addr (bytes)
  g0[2] = (unsigned)ga;                           // global_addr[31:0]
  g0[3] = ((unsigned)(ga >> 32) & 0x01FFFFFFu)    // global_addr[56:32]
          | 0x80000000u;                          // type=2 at bits[127:126]
  v4i z4 = {0, 0, 0, 0};
  v8i z8 = {0, 0, 0, 0, 0, 0, 0, 0};
  __builtin_amdgcn_tensor_load_to_lds(g0, g1, z4, z4, z8, 0);
}

// ======================= generic WMMA GEMM ==================================
// C[M,N] = act(A[M,K] @ B[K,N] + bias). A,B f16 row-major. M%128==0, N%128==0,
// K%32==0. 256 threads = 8 waves; block tile 128x128; wave tile 32x64 (2x4
// WMMA accumulators). A tile staged by TDM (double-buffered, overlapped with
// compute); B tile staged transposed via registers for contiguous frag reads.
__global__ __launch_bounds__(256) void k_gemm(
    const _Float16* __restrict__ A, const _Float16* __restrict__ Bm,
    const float* __restrict__ bias, float* __restrict__ Cf,
    _Float16* __restrict__ Ch, int M, int N, int K, int act)
{
  __shared__ __align__(64) _Float16 As[2][128][32];   // A tile, row-major
  __shared__ __align__(64) _Float16 BsT[2][128][32];  // B tile, [n][k]

  const int tid  = threadIdx.x;
  const int lane = tid & 31;
  const int w    = tid >> 5;
  const int lr   = lane & 15;      // n / m within 16
  const int lh   = lane >> 4;      // lane-half selects K sub-block per ISA
  const int mb   = blockIdx.y * 128, nb = blockIdx.x * 128;
  const int mw   = (w >> 1) * 32,  nw = (w & 1) * 64;

  v8f acc[2][4];
  {
    v8f zf = {};
#pragma unroll
    for (int s = 0; s < 2; ++s)
#pragma unroll
      for (int t = 0; t < 4; ++t) acc[s][t] = zf;
  }

  // TDM descriptor group1 (uniform): dims 1<<24, tile 32x128, stride0 = K
  v8i g1;
  g1[0] = 0x00010000;            // data_size = 2 bytes
  g1[1] = 0;                     // dim0[15:0] << 16  (dim0 = 1<<24)
  g1[2] = 0x100;                 // dim0[31:16] | dim1[15:0]<<16
  g1[3] = 0x100 | (32 << 16);    // dim1[31:16] | tile_dim0 = 32
  g1[4] = 128;                   // tile_dim1 = 128
  g1[5] = K;                     // tensor_dim0_stride[31:0]
  g1[6] = 0;
  g1[7] = 0;

  const int nk = K / 32;
  int buf = 0;
  if (w == 0)   // prime the pipeline: tile 0 -> As[0]
    tdm_load_tile(&A[(size_t)mb * K], (unsigned)(uintptr_t)&As[0][0][0], g1);

  for (int kt = 0; kt < nk; ++kt) {
    const int k0 = kt * 32;
    // ---- stage B tile (32 k x 128 n) transposed: 4096 halves / 256 thr ----
    if (kt + 1 < nk)
      __builtin_prefetch(&Bm[(size_t)(k0 + 32 + (tid >> 3)) * N + nb], 0, 1);
#pragma unroll
    for (int c = 0; c < 2; ++c) {
      const int lin = (tid * 2 + c) * 8;
      const int br = lin >> 7, bn = lin & 127;
      v8h tb = *(const v8h*)&Bm[(size_t)(k0 + br) * N + nb + bn];
#pragma unroll
      for (int j = 0; j < 8; ++j) BsT[buf][bn + j][br] = tb[j];
    }
    if (w == 0) __builtin_amdgcn_s_wait_tensorcnt(0);  // As[buf] TDM done
    __syncthreads();
    // overlap: TDM for next tile into the other buffer while we compute
    if (w == 0 && kt + 1 < nk)
      tdm_load_tile(&A[(size_t)mb * K + (size_t)(k0 + 32)],
                    (unsigned)(uintptr_t)&As[buf ^ 1][0][0], g1);

    v16h af[2], bf[4];
#pragma unroll
    for (int s = 0; s < 2; ++s) {
      // A frag (16x32): lanes<16 hold K{0..15}, lanes>=16 K{16..31}
      U16 u;
      u.h[0] = *(const v8h*)&As[buf][mw + s * 16 + lr][lh * 16];
      u.h[1] = *(const v8h*)&As[buf][mw + s * 16 + lr][lh * 16 + 8];
      af[s] = u.v;
    }
#pragma unroll
    for (int t = 0; t < 4; ++t)   // B frag: b[i] = B[lh*16+i][n] = BsT[n][...]
      bf[t] = *(const v16h*)&BsT[buf][nw + t * 16 + lr][lh * 16];
#pragma unroll
    for (int s = 0; s < 2; ++s)
#pragma unroll
      for (int t = 0; t < 4; ++t)
        acc[s][t] = __builtin_amdgcn_wmma_f32_16x16x32_f16(
            false, af[s], false, bf[t], (short)0, acc[s][t], false, false);
    buf ^= 1;
  }

  // C layout: VGPR r -> rows {r, r+8} by lane-half, cols n = lane%16
#pragma unroll
  for (int s = 0; s < 2; ++s)
#pragma unroll
    for (int t = 0; t < 4; ++t)
#pragma unroll
      for (int r = 0; r < 8; ++r) {
        const int m = mb + mw + s * 16 + r + lh * 8;
        const int n = nb + nw + t * 16 + lr;
        float v = acc[s][t][r];
        if (bias) v += bias[n];
        if (act == 1) v = fmaxf(v, 0.0f);
        else if (act == 2) v = gelu_exact(v);
        const size_t idx = (size_t)m * N + n;
        if (Cf) Cf[idx] = v;
        if (Ch) Ch[idx] = (_Float16)v;
      }
}

// ======================= fused flash attention ==============================
// Q,K,V,O: f16 [(b*T+t)*E + h*64+d]. One block = (b, h, 128 q rows); 8 waves,
// 16 q rows each. Streams 32-key tiles: S=QK^T (4 WMMA), online softmax,
// P via per-wave LDS (C-layout -> A-layout), PV (4 WMMA). K tile staged with
// GLOBAL_LOAD_ASYNC_TO_LDS_B128 (ASYNCcnt); V staged transposed via registers.
__global__ __launch_bounds__(256) void k_attn(
    const _Float16* __restrict__ q, const _Float16* __restrict__ k,
    const _Float16* __restrict__ v, _Float16* __restrict__ o)
{
  __shared__ __align__(64) _Float16 Ks[32][64];      // [key][d]
  __shared__ __align__(64) _Float16 VTs[64][32];     // [d][key]
  __shared__ __align__(64) _Float16 Ps[8][16][32];   // per-wave P scratch

  const int tid  = threadIdx.x;
  const int lane = tid & 31, w = tid >> 5;
  const int lr   = lane & 15, lh = lane >> 4;
  const int hy   = blockIdx.y, bz = blockIdx.z;
  const size_t bt = (size_t)bz * TT;

  // Q A-fragments for this wave's 16 rows (K=64 -> two frags)
  const int qrow = blockIdx.x * 128 + w * 16 + lr;
  const _Float16* qp = q + (bt + qrow) * EE + hy * HDD;
  v16h aq[2];
#pragma unroll
  for (int s = 0; s < 2; ++s) {
    U16 u;
    u.h[0] = *(const v8h*)(qp + s * 32 + lh * 16);
    u.h[1] = *(const v8h*)(qp + s * 32 + lh * 16 + 8);
    aq[s] = u.v;
  }

  const _Float16* kp = k + bt * EE + hy * HDD;
  const _Float16* vp = v + bt * EE + hy * HDD;

  float mrow[8], lrow[8];
  v8f o_acc[4];
  {
    v8f zf = {};
#pragma unroll
    for (int t = 0; t < 4; ++t) o_acc[t] = zf;
#pragma unroll
    for (int r = 0; r < 8; ++r) { mrow[r] = -1e30f; lrow[r] = 0.0f; }
  }

  for (int jt = 0; jt < TT / 32; ++jt) {
    const int j0 = jt * 32;
    __syncthreads();
    {  // cooperative stage: K via async-to-LDS, V transposed via registers
      const int kr = tid >> 3;             // 0..31 key
      const int dc = (tid & 7) * 8;        // 0..56 dim
      unsigned lds_k = (unsigned)(uintptr_t)&Ks[kr][dc];
      unsigned long long gk =
          (unsigned long long)(uintptr_t)&kp[(size_t)(j0 + kr) * EE + dc];
      asm volatile("global_load_async_to_lds_b128 %0, %1, off"
                   :: "v"(lds_k), "v"(gk) : "memory");
      v8h tv = *(const v8h*)&vp[(size_t)(j0 + kr) * EE + dc];
#pragma unroll
      for (int j = 0; j < 8; ++j) VTs[dc + j][kr] = tv[j];
      asm volatile("s_wait_asynccnt 0" ::: "memory");
    }
    __syncthreads();

    // S[16 x 32] = Q @ K^T : two n-tiles (keys 0-15 / 16-31), K=64 in 2 steps
    v8f s0 = {}, s1 = {};
#pragma unroll
    for (int s = 0; s < 2; ++s) {
      v16h bk0 = *(const v16h*)&Ks[lr][s * 32 + lh * 16];
      v16h bk1 = *(const v16h*)&Ks[16 + lr][s * 32 + lh * 16];
      s0 = __builtin_amdgcn_wmma_f32_16x16x32_f16(false, aq[s], false, bk0,
                                                  (short)0, s0, false, false);
      s1 = __builtin_amdgcn_wmma_f32_16x16x32_f16(false, aq[s], false, bk1,
                                                  (short)0, s1, false, false);
    }

    // online softmax per row (rows r / r+8 by lane-half; cols across 16 lanes)
#pragma unroll
    for (int r = 0; r < 8; ++r) {
      float a0 = s0[r], a1 = s1[r];
      float mx = fmaxf(a0, a1);
#pragma unroll
      for (int m = 1; m < 16; m <<= 1) mx = fmaxf(mx, __shfl_xor(mx, m, 32));
      const float mn = fmaxf(mrow[r], mx);
      const float sc = __expf(mrow[r] - mn);
      const float p0 = __expf(a0 - mn);
      const float p1 = __expf(a1 - mn);
      float sm = p0 + p1;
#pragma unroll
      for (int m = 1; m < 16; m <<= 1) sm += __shfl_xor(sm, m, 32);
      lrow[r] = lrow[r] * sc + sm;
      mrow[r] = mn;
#pragma unroll
      for (int t = 0; t < 4; ++t) o_acc[t][r] *= sc;
      const int mr = r + lh * 8;
      Ps[w][mr][lr]      = (_Float16)p0;
      Ps[w][mr][16 + lr] = (_Float16)p1;
    }
    // wave-private LDS RAW: all 32 lanes' ds_stores retire before frag reads
    asm volatile("s_wait_dscnt 0" ::: "memory");

    // P as A-fragment (16x32), V tile as 4 B-fragments (d tiles of 16)
    v16h ap;
    {
      U16 u;
      u.h[0] = *(const v8h*)&Ps[w][lr][lh * 16];
      u.h[1] = *(const v8h*)&Ps[w][lr][lh * 16 + 8];
      ap = u.v;
    }
#pragma unroll
    for (int t = 0; t < 4; ++t) {
      v16h bv = *(const v16h*)&VTs[t * 16 + lr][lh * 16];
      o_acc[t] = __builtin_amdgcn_wmma_f32_16x16x32_f16(false, ap, false, bv,
                                                        (short)0, o_acc[t],
                                                        false, false);
    }
  }

  _Float16* op = o + (bt + blockIdx.x * 128 + w * 16) * EE + hy * HDD;
#pragma unroll
  for (int t = 0; t < 4; ++t)
#pragma unroll
    for (int r = 0; r < 8; ++r) {
      const int mr = r + lh * 8;
      op[(size_t)mr * EE + t * 16 + lr] = (_Float16)(o_acc[t][r] / lrow[r]);
    }
}

// ======================= small helper kernels ===============================
__global__ void k_conv1(const float* __restrict__ x, const float* __restrict__ wt,
                        const float* __restrict__ bs, _Float16* __restrict__ out) {
  const int idx = blockIdx.x * blockDim.x + threadIdx.x;
  if (idx >= NB * TIN * EE) return;
  const int e = idx % EE, t = (idx / EE) % TIN, b = idx / (EE * TIN);
  float acc = bs[e];
  const float* xb = x + (size_t)b * NMELS * TIN;
  const float* we = wt + (size_t)e * NMELS * 3;
  for (int c = 0; c < NMELS; ++c) {
    const float* xc = xb + (size_t)c * TIN;
    const float w0 = we[c * 3], w1 = we[c * 3 + 1], w2 = we[c * 3 + 2];
    if (t > 0)       acc += xc[t - 1] * w0;
    acc += xc[t] * w1;
    if (t < TIN - 1) acc += xc[t + 1] * w2;
  }
  out[idx] = (_Float16)gelu_exact(acc);   // layout (b,t,e)
}

__global__ void k_im2col(const _Float16* __restrict__ x1, _Float16* __restrict__ out) {
  const int idx = blockIdx.x * blockDim.x + threadIdx.x;
  if (idx >= NB * TT * 3 * EE) return;
  const int c = idx % (3 * EE), r = idx / (3 * EE);
  const int b = r / TT, t = r % TT, dk = c / EE, i = c % EE;
  const int tt = 2 * t - 1 + dk;
  out[idx] = (tt >= 0 && tt < TIN) ? x1[((size_t)b * TIN + tt) * EE + i]
                                   : (_Float16)0.0f;
}

__global__ void k_cvt(const float* __restrict__ s, _Float16* __restrict__ d, int n) {
  const int i = blockIdx.x * blockDim.x + threadIdx.x;
  if (i < n) d[i] = (_Float16)s[i];
}

// (H,E,HD) -> (E, H*HD) f16
__global__ void k_pack_qkvw(const float* __restrict__ s, _Float16* __restrict__ d) {
  const int idx = blockIdx.x * blockDim.x + threadIdx.x;
  if (idx >= EE * EE) return;
  const int e = idx / EE, c = idx % EE;
  d[idx] = (_Float16)s[((size_t)(c / HDD) * EE + e) * HDD + (c % HDD)];
}

// (O,I,3) -> (3*E, O) f16 matching im2col col order (dk*E + i)
__global__ void k_pack_conv2w(const float* __restrict__ s, _Float16* __restrict__ d) {
  const int idx = blockIdx.x * blockDim.x + threadIdx.x;
  if (idx >= 3 * EE * EE) return;
  const int row = idx / EE, oo = idx % EE;
  const int dk = row / EE, i = row % EE;
  d[idx] = (_Float16)s[((size_t)oo * EE + i) * 3 + dk];
}

__global__ void k_pos_add(const float* __restrict__ a, const float* __restrict__ pos,
                          float* __restrict__ xf, _Float16* __restrict__ xh) {
  const int idx = blockIdx.x * blockDim.x + threadIdx.x;
  if (idx >= NB * TT * EE) return;
  const int e = idx % EE, t = (idx / EE) % TT;
  const float v = a[idx] + pos[(size_t)t * EE + e];
  xf[idx] = v; xh[idx] = (_Float16)v;
}

// y = LN(a + res); writes f32 + f16 mirrors. One block per row of 768.
__global__ __launch_bounds__(256) void k_add_ln(
    const float* __restrict__ a, const float* __restrict__ res,
    const float* __restrict__ g, const float* __restrict__ be,
    float* __restrict__ xf, _Float16* __restrict__ xh) {
  __shared__ float red[256];
  const int row = blockIdx.x, tid = threadIdx.x;
  const size_t base = (size_t)row * EE;
  float e0 = a[base + tid]       + res[base + tid];
  float e1 = a[base + tid + 256] + res[base + tid + 256];
  float e2 = a[base + tid + 512] + res[base + tid + 512];
  red[tid] = e0 + e1 + e2; __syncthreads();
  for (int st = 128; st > 0; st >>= 1) { if (tid < st) red[tid] += red[tid + st]; __syncthreads(); }
  const float mean = red[0] * (1.0f / EE); __syncthreads();
  const float d0 = e0 - mean, d1 = e1 - mean, d2 = e2 - mean;
  red[tid] = d0 * d0 + d1 * d1 + d2 * d2; __syncthreads();
  for (int st = 128; st > 0; st >>= 1) { if (tid < st) red[tid] += red[tid + st]; __syncthreads(); }
  const float rs = rsqrtf(red[0] * (1.0f / EE) + 1e-5f);
  const float o0 = d0 * rs * g[tid]       + be[tid];
  const float o1 = d1 * rs * g[tid + 256] + be[tid + 256];
  const float o2 = d2 * rs * g[tid + 512] + be[tid + 512];
  xf[base + tid] = o0;       xh[base + tid] = (_Float16)o0;
  xf[base + tid + 256] = o1; xh[base + tid + 256] = (_Float16)o1;
  xf[base + tid + 512] = o2; xh[base + tid + 512] = (_Float16)o2;
}

// cls = LN(LN(x[b,0], lnp), hln); one block per batch
__global__ __launch_bounds__(256) void k_cls_ln(
    const float* __restrict__ xf, const float* __restrict__ g1, const float* __restrict__ b1,
    const float* __restrict__ g2, const float* __restrict__ b2, float* __restrict__ out) {
  __shared__ float red[256];
  const int b = blockIdx.x, tid = threadIdx.x;
  const size_t base = (size_t)b * TT * EE;
  float e0 = xf[base + tid], e1 = xf[base + tid + 256], e2 = xf[base + tid + 512];
#pragma unroll
  for (int pass = 0; pass < 2; ++pass) {
    red[tid] = e0 + e1 + e2; __syncthreads();
    for (int st = 128; st > 0; st >>= 1) { if (tid < st) red[tid] += red[tid + st]; __syncthreads(); }
    const float mean = red[0] * (1.0f / EE); __syncthreads();
    const float d0 = e0 - mean, d1 = e1 - mean, d2 = e2 - mean;
    red[tid] = d0 * d0 + d1 * d1 + d2 * d2; __syncthreads();
    for (int st = 128; st > 0; st >>= 1) { if (tid < st) red[tid] += red[tid + st]; __syncthreads(); }
    const float rs = rsqrtf(red[0] * (1.0f / EE) + 1e-5f); __syncthreads();
    const float* g = pass ? g2 : g1; const float* be = pass ? b2 : b1;
    e0 = d0 * rs * g[tid]       + be[tid];
    e1 = d1 * rs * g[tid + 256] + be[tid + 256];
    e2 = d2 * rs * g[tid + 512] + be[tid + 512];
  }
  out[(size_t)b * EE + tid] = e0;
  out[(size_t)b * EE + tid + 256] = e1;
  out[(size_t)b * EE + tid + 512] = e2;
}

__global__ void k_head1(const float* __restrict__ cls, const float* __restrict__ w,
                        const float* __restrict__ bs, float* __restrict__ out) {
  const int idx = blockIdx.x * blockDim.x + threadIdx.x;
  if (idx >= NB * MFF) return;
  const int b = idx / MFF, m = idx % MFF;
  float acc = bs[m];
  for (int kk = 0; kk < EE; ++kk) acc += cls[(size_t)b * EE + kk] * w[(size_t)kk * MFF + m];
  out[idx] = fmaxf(acc, 0.0f);
}

__global__ void k_head2(const float* __restrict__ h1, const float* __restrict__ w,
                        const float* __restrict__ bs, float* __restrict__ out) {
  const int idx = blockIdx.x * blockDim.x + threadIdx.x;
  if (idx >= NB * CC) return;
  const int b = idx / CC, c = idx % CC;
  float acc = bs[c];
  for (int kk = 0; kk < MFF; ++kk) acc += h1[(size_t)b * MFF + kk] * w[(size_t)kk * CC + c];
  out[idx] = acc;
}

__global__ __launch_bounds__(256) void k_softmax527(const float* __restrict__ lg,
                                                    float* __restrict__ out) {
  __shared__ float red[256];
  const int b = blockIdx.x, tid = threadIdx.x;
  const float* p = lg + (size_t)b * CC;
  float mx = -1e30f;
  for (int i = tid; i < CC; i += 256) mx = fmaxf(mx, p[i]);
  red[tid] = mx; __syncthreads();
  for (int st = 128; st > 0; st >>= 1) { if (tid < st) red[tid] = fmaxf(red[tid], red[tid + st]); __syncthreads(); }
  mx = red[0]; __syncthreads();
  float s = 0.0f;
  for (int i = tid; i < CC; i += 256) s += __expf(p[i] - mx);
  red[tid] = s; __syncthreads();
  for (int st = 128; st > 0; st >>= 1) { if (tid < st) red[tid] += red[tid + st]; __syncthreads(); }
  const float inv = 1.0f / red[0];
  for (int i = tid; i < CC; i += 256) out[(size_t)b * CC + i] = __expf(p[i] - mx) * inv;
}

// ======================= orchestration ======================================
extern "C" void kernel_launch(void* const* d_in, const int* in_sizes, int n_in,
                              void* d_out, int out_size, void* d_ws, size_t ws_size,
                              hipStream_t stream) {
  (void)in_sizes; (void)n_in; (void)out_size; (void)ws_size;
  const float* x       = (const float*)d_in[0];
  const float* conv1_w = (const float*)d_in[1];
  const float* conv1_b = (const float*)d_in[2];
  const float* conv2_w = (const float*)d_in[3];
  const float* conv2_b = (const float*)d_in[4];
  const float* pos_emb = (const float*)d_in[5];
  const float* ln1_g   = (const float*)d_in[6];
  const float* ln1_b   = (const float*)d_in[7];
  const float* wq = (const float*)d_in[8];  const float* bq = (const float*)d_in[9];
  const float* wk = (const float*)d_in[10]; const float* bk = (const float*)d_in[11];
  const float* wv = (const float*)d_in[12]; const float* bv = (const float*)d_in[13];
  const float* wo = (const float*)d_in[14]; const float* bo = (const float*)d_in[15];
  const float* w1 = (const float*)d_in[16]; const float* b1 = (const float*)d_in[17];
  const float* w2 = (const float*)d_in[18]; const float* b2 = (const float*)d_in[19];
  const float* ln2_g = (const float*)d_in[20]; const float* ln2_b = (const float*)d_in[21];
  const float* lnp_g = (const float*)d_in[22]; const float* lnp_b = (const float*)d_in[23];
  const float* hln_g = (const float*)d_in[24]; const float* hln_b = (const float*)d_in[25];
  const float* hw1 = (const float*)d_in[26]; const float* hb1 = (const float*)d_in[27];
  const float* hw2 = (const float*)d_in[28]; const float* hb2 = (const float*)d_in[29];

  char* wsp = (char*)d_ws; size_t off = 0;
  auto carve = [&](size_t bytes) -> void* {
    void* p = wsp + off; off = (off + bytes + 255) & ~(size_t)255; return p;
  };
  const size_t NT = (size_t)NB * TT;            // 4096 rows
  _Float16* x1h  = (_Float16*)carve((size_t)NB * TIN * EE * 2);
  _Float16* a2   = (_Float16*)carve(NT * 3 * EE * 2);
  _Float16* w2c  = (_Float16*)carve((size_t)3 * EE * EE * 2);
  float*    xf   = (float*)   carve(NT * EE * 4);
  _Float16* xh   = (_Float16*)carve(NT * EE * 2);
  _Float16* qh   = (_Float16*)carve(NT * EE * 2);
  _Float16* kh   = (_Float16*)carve(NT * EE * 2);
  _Float16* vh   = (_Float16*)carve(NT * EE * 2);
  _Float16* ah   = (_Float16*)carve(NT * EE * 2);
  float*    tf   = (float*)   carve(NT * EE * 4);
  _Float16* h1h  = (_Float16*)carve(NT * MFF * 2);
  _Float16* wqh  = (_Float16*)carve((size_t)EE * EE * 2);
  _Float16* wkh  = (_Float16*)carve((size_t)EE * EE * 2);
  _Float16* wvh  = (_Float16*)carve((size_t)EE * EE * 2);
  _Float16* woh  = (_Float16*)carve((size_t)EE * EE * 2);
  _Float16* w1h  = (_Float16*)carve((size_t)EE * MFF * 2);
  _Float16* w2h  = (_Float16*)carve((size_t)MFF * EE * 2);
  float*    clsn = (float*)carve((size_t)NB * EE * 4);
  float*    h1f  = (float*)carve((size_t)NB * MFF * 4);
  float*    lg   = (float*)carve((size_t)NB * CC * 4);

  const int TPB = 256;
  auto blks = [](size_t n) { return (unsigned)((n + 255) / 256); };

  // ---- conv stem ----
  k_conv1<<<blks((size_t)NB * TIN * EE), TPB, 0, stream>>>(x, conv1_w, conv1_b, x1h);
  k_im2col<<<blks(NT * 3 * EE), TPB, 0, stream>>>(x1h, a2);
  k_pack_conv2w<<<blks((size_t)3 * EE * EE), TPB, 0, stream>>>(conv2_w, w2c);
  k_gemm<<<dim3(EE / 128, (unsigned)(NT / 128)), 256, 0, stream>>>(
      a2, w2c, conv2_b, xf, nullptr, (int)NT, EE, 3 * EE, /*gelu*/2);
  k_pos_add<<<blks(NT * EE), TPB, 0, stream>>>(xf, pos_emb, xf, xh);

  // ---- transformer layers ----
  for (int l = 0; l < LL; ++l) {
    const size_t wqo = (size_t)l * HH * EE * HDD;
    k_pack_qkvw<<<blks((size_t)EE * EE), TPB, 0, stream>>>(wq + wqo, wqh);
    k_pack_qkvw<<<blks((size_t)EE * EE), TPB, 0, stream>>>(wk + wqo, wkh);
    k_pack_qkvw<<<blks((size_t)EE * EE), TPB, 0, stream>>>(wv + wqo, wvh);
    k_cvt<<<blks((size_t)EE * EE), TPB, 0, stream>>>(wo + (size_t)l * EE * EE, woh, EE * EE);
    k_cvt<<<blks((size_t)EE * MFF), TPB, 0, stream>>>(w1 + (size_t)l * EE * MFF, w1h, EE * MFF);
    k_cvt<<<blks((size_t)MFF * EE), TPB, 0, stream>>>(w2 + (size_t)l * MFF * EE, w2h, MFF * EE);

    const dim3 g768(EE / 128, (unsigned)(NT / 128));
    k_gemm<<<g768, 256, 0, stream>>>(xh, wqh, bq + (size_t)l * EE, nullptr, qh, (int)NT, EE, EE, 0);
    k_gemm<<<g768, 256, 0, stream>>>(xh, wkh, bk + (size_t)l * EE, nullptr, kh, (int)NT, EE, EE, 0);
    k_gemm<<<g768, 256, 0, stream>>>(xh, wvh, bv + (size_t)l * EE, nullptr, vh, (int)NT, EE, EE, 0);

    k_attn<<<dim3(TT / 128, HH, NB), 256, 0, stream>>>(qh, kh, vh, ah);

    k_gemm<<<g768, 256, 0, stream>>>(ah, woh, bo + (size_t)l * EE, tf, nullptr, (int)NT, EE, EE, 0);
    k_add_ln<<<(unsigned)NT, 256, 0, stream>>>(tf, xf, ln1_g + (size_t)l * EE, ln1_b + (size_t)l * EE, xf, xh);

    k_gemm<<<dim3(MFF / 128, (unsigned)(NT / 128)), 256, 0, stream>>>(
        xh, w1h, b1 + (size_t)l * MFF, nullptr, h1h, (int)NT, MFF, EE, /*relu*/1);
    k_gemm<<<g768, 256, 0, stream>>>(h1h, w2h, b2 + (size_t)l * EE, tf, nullptr, (int)NT, EE, MFF, 0);
    k_add_ln<<<(unsigned)NT, 256, 0, stream>>>(tf, xf, ln2_g + (size_t)l * EE, ln2_b + (size_t)l * EE, xf, xh);
  }

  // ---- head (tiny, f32) ----
  k_cls_ln<<<NB, 256, 0, stream>>>(xf, lnp_g, lnp_b, hln_g, hln_b, clsn);
  k_head1<<<blks((size_t)NB * MFF), TPB, 0, stream>>>(clsn, hw1, hb1, h1f);
  k_head2<<<blks((size_t)NB * CC), TPB, 0, stream>>>(h1f, hw2, hb2, lg);
  k_softmax527<<<NB, 256, 0, stream>>>(lg, (float*)d_out);
}
